// TSPPruningGNN_35321811042630
// MI455X (gfx1250) — matrix operand order repaired and checked by reading, not compiled, not run on hardware
//
#include <hip/hip_runtime.h>
#include <hip/hip_bf16.h>

typedef __attribute__((ext_vector_type(16))) _Float16 v16h;
typedef __attribute__((ext_vector_type(8)))  _Float16 v8h;
typedef __attribute__((ext_vector_type(4)))  _Float16 v4h;
typedef __attribute__((ext_vector_type(8)))  float    v8f;
typedef __attribute__((ext_vector_type(4)))  float    v4f;

#define WMMA_F16F32(a, b, c) \
    __builtin_amdgcn_wmma_f32_16x16x32_f16(false, (a), false, (b), (short)0, (c), false, false)

static inline unsigned cdiv_u(long long a, long long b) { return (unsigned)((a + b - 1) / b); }
static inline unsigned umin_u(unsigned a, unsigned b) { return a < b ? a : b; }

// ---------------- degree / normalization ----------------
__global__ void k_deg_init(float* __restrict__ deg, int N) {
    int i = blockIdx.x * blockDim.x + threadIdx.x;
    if (i < N) deg[i] = 1.0f;  // self-loop
}

__global__ void k_deg_count(const int* __restrict__ row, float* __restrict__ deg, int E) {
    int i = blockIdx.x * blockDim.x + threadIdx.x;
    if (i < E) atomicAdd(&deg[row[i]], 1.0f);
}

__global__ void k_dinv(float* __restrict__ deg, int N) {
    int i = blockIdx.x * blockDim.x + threadIdx.x;
    if (i < N) deg[i] = rsqrtf(deg[i]);  // deg >= 1 always
}

// ---------------- weights -> f16 (once) ----------------
__global__ void k_wcvt(const float* __restrict__ W2, const float* __restrict__ Wc1,
                       _Float16* __restrict__ W2h, _Float16* __restrict__ Wc1h) {
    int t = blockIdx.x * blockDim.x + threadIdx.x;
    if (t < 64 * 32) W2h[t] = (_Float16)W2[t];
    if (t < 64 * 16) Wc1h[t] = (_Float16)Wc1[t];
}

// ------- fused: xh1 = x@W1 ; agg1 = dinv^2 * xh1 (self-loop term) -------
__global__ void k_xw1_selfloop(const float* __restrict__ x, const float* __restrict__ W1,
                               const float* __restrict__ dinv,
                               float* __restrict__ xh1, float* __restrict__ agg1, int N) {
    int t = blockIdx.x * blockDim.x + threadIdx.x;
    if (t >= N * 64) return;
    int n = t >> 6, o = t & 63;
    const float* xr = x + (long)n * 9;
    float s = 0.0f;
#pragma unroll
    for (int k = 0; k < 9; ++k) s += xr[k] * W1[k * 64 + o];
    xh1[t] = s;
    float d = dinv[n];
    agg1[t] = d * d * s;
}

// ------- scatter, 64 features: one wave per edge; uniform idx via SMEM -------
__global__ void k_scatter64(const int* __restrict__ row, const int* __restrict__ col,
                            const float* __restrict__ dinv, const float* __restrict__ xh,
                            float* __restrict__ agg, int E) {
    int e = __builtin_amdgcn_readfirstlane((blockIdx.x * blockDim.x + threadIdx.x) >> 5);
    if (e >= E) return;
    int l = threadIdx.x & 31;
    int r = row[e], c = col[e];           // scalar loads (wave-uniform)
    float norm = dinv[r] * dinv[c];
    const float* src = xh + (long)c * 64;  // SGPR base
    float* dst = agg + (long)r * 64;       // SGPR base
    atomicAdd(dst + l,      norm * src[l]);
    atomicAdd(dst + l + 32, norm * src[l + 32]);
}

// ------- scatter, 32 features: one wave per edge; uniform idx via SMEM -------
__global__ void k_scatter32(const int* __restrict__ row, const int* __restrict__ col,
                            const float* __restrict__ dinv, const float* __restrict__ xh,
                            float* __restrict__ agg, int E) {
    int e = __builtin_amdgcn_readfirstlane((blockIdx.x * blockDim.x + threadIdx.x) >> 5);
    if (e >= E) return;
    int l = threadIdx.x & 31;
    int r = row[e], c = col[e];
    float norm = dinv[r] * dinv[c];
    atomicAdd(agg + (long)r * 32 + l, norm * xh[(long)c * 32 + l]);
}

// ---------------- h1 = relu(agg1 + b1) -> f16, 4 elems/thread ----------------
__global__ void k_relu_bias_f16(const float* __restrict__ agg, const float* __restrict__ b,
                                _Float16* __restrict__ h, int total4) {
    int t4 = blockIdx.x * blockDim.x + threadIdx.x;
    if (t4 >= total4) return;
    int t = t4 * 4;
    v4f a = *(const v4f*)(agg + t);
    int o = t & 63;
    v4h r;
#pragma unroll
    for (int i = 0; i < 4; ++i) r[i] = (_Float16)fmaxf(a[i] + b[o + i], 0.0f);
    *(v4h*)(h + t) = r;
}

// -------- xh2 = h1f16[N,64] @ W2[64,32] via WMMA (persistent, pipelined) -----
// Fused epilogue: agg2 = dinv^2 * xh2 (self-loop term of conv2).
__global__ void k_gemm2_wmma(const _Float16* __restrict__ h1, const _Float16* __restrict__ W2h,
                             const float* __restrict__ dinv,
                             float* __restrict__ xh2, float* __restrict__ agg2,
                             int N, int ntiles) {
    int lane   = threadIdx.x & 31;
    int wave0  = (blockIdx.x * blockDim.x + threadIdx.x) >> 5;
    int nwaves = (gridDim.x * blockDim.x) >> 5;
    if (wave0 >= ntiles) return;  // wave-uniform

    int m    = lane & 15;
    int half = lane >> 4;
    int ncol = lane & 15;
    int kb   = half * 16;

    // hoisted B operands (f16, once per wave)
    v16h b00, b01, b10, b11;
#pragma unroll
    for (int i = 0; i < 16; ++i) {
        b00[i] = W2h[(kb + i) * 32 + ncol];
        b01[i] = W2h[(kb + i) * 32 + 16 + ncol];
        b10[i] = W2h[(32 + kb + i) * 32 + ncol];
        b11[i] = W2h[(32 + kb + i) * 32 + 16 + ncol];
    }

    // prologue: issue A loads for first tile
    int node0 = wave0 * 16 + m;
    if (node0 >= N) node0 = N - 1;
    const v8h* hr8 = (const v8h*)(h1 + (long)node0 * 64);
    v8h l0 = hr8[2 * half], l1 = hr8[2 * half + 1], l2 = hr8[2 * half + 4], l3 = hr8[2 * half + 5];

    for (int tile = wave0; tile < ntiles; tile += nwaves) {
        // speculatively issue next tile's A loads (clamp covers tn >= ntiles:
        // tn*16+m >= N then). Younger loads; in-order return means the WMMA
        // wait below only covers current-tile loads.
        int noden = (tile + nwaves) * 16 + m;
        if (noden >= N) noden = N - 1;
        const v8h* hn = (const v8h*)(h1 + (long)noden * 64);
        v8h n0 = hn[2 * half], n1 = hn[2 * half + 1], n2 = hn[2 * half + 4], n3 = hn[2 * half + 5];

        v16h a0, a1;
#pragma unroll
        for (int i = 0; i < 8; ++i) {
            a0[i] = l0[i]; a0[8 + i] = l1[i];
            a1[i] = l2[i]; a1[8 + i] = l3[i];
        }

        v8f c0 = {}, c1 = {};
        c0 = WMMA_F16F32(a0, b00, c0);
        c0 = WMMA_F16F32(a1, b10, c0);
        c1 = WMMA_F16F32(a0, b01, c1);
        c1 = WMMA_F16F32(a1, b11, c1);

        int rbase = tile * 16 + half * 8;
        float* px = xh2  + (long)rbase * 32 + ncol;  // one base per tile;
        float* pa = agg2 + (long)rbase * 32 + ncol;  // rows at +v*128B imm offsets
        if (tile * 16 + 16 <= N) {
            // full-tile fast path: dinv preloaded as 2x v4f, unconditional stores
            const v4f* dv = (const v4f*)(dinv + rbase);
            v4f d0 = dv[0], d1 = dv[1];
            float dd[8];
#pragma unroll
            for (int v = 0; v < 4; ++v) { dd[v] = d0[v] * d0[v]; dd[4 + v] = d1[v] * d1[v]; }
#pragma unroll
            for (int v = 0; v < 8; ++v) {
                px[v * 32]      = c0[v];
                px[v * 32 + 16] = c1[v];
                pa[v * 32]      = dd[v] * c0[v];
                pa[v * 32 + 16] = dd[v] * c1[v];
            }
        } else {
#pragma unroll
            for (int v = 0; v < 8; ++v) {
                if (rbase + v < N) {
                    float d  = dinv[rbase + v];
                    float dd = d * d;
                    px[v * 32]      = c0[v];
                    px[v * 32 + 16] = c1[v];
                    pa[v * 32]      = dd * c0[v];
                    pa[v * 32 + 16] = dd * c1[v];
                }
            }
        }

        l0 = n0; l1 = n1; l2 = n2; l3 = n3;
    }
}

// ---- edge MLP: [h2[row] || h2[col]] @ Wc1 -> relu -> @Wc2 -> sigmoid ----
// Persistent waves; indices register-pipelined 2 tiles ahead; rows prefetched
// (global_prefetch_b8) ~2 iterations before their gather; +b2 fused in gather.
__global__ void k_edge_wmma(const float* __restrict__ h2,
                            const int* __restrict__ row, const int* __restrict__ col,
                            const _Float16* __restrict__ Wc1h, const float* __restrict__ bc1,
                            const float* __restrict__ Wc2, const float* __restrict__ bc2,
                            const float* __restrict__ b2,
                            float* __restrict__ out, int E, int ntiles) {
    int lane   = threadIdx.x & 31;
    int wave0  = (blockIdx.x * blockDim.x + threadIdx.x) >> 5;
    int nwaves = (gridDim.x * blockDim.x) >> 5;
    if (wave0 >= ntiles) return;  // wave-uniform

    int m    = lane & 15;
    int half = lane >> 4;
    int k0   = half * 8;
    int ncol = lane & 15;
    int kb   = half * 16;

    // hoisted B operands
    v16h bR, bC;
#pragma unroll
    for (int i = 0; i < 16; ++i) {
        bR[i] = Wc1h[(i < 8 ? k0 + i : k0 + 8 + i) * 16 + ncol];
        bC[i] = Wc1h[(32 + (i < 8 ? k0 + i : k0 + 8 + i)) * 16 + ncol];
    }
    float bb  = bc1[ncol];
    float w2  = Wc2[ncol];
    float b2s = bc2[0];
    float badd[16];
#pragma unroll
    for (int i = 0; i < 8; ++i) {
        badd[i]     = b2[k0 + i];
        badd[8 + i] = b2[k0 + 16 + i];
    }

    // index pipeline (branch-free: clamp covers tiles past the end)
    long e0 = (long)wave0 * 16 + m;
    if (e0 >= E) e0 = E - 1;
    int rcur = row[e0], ccur = col[e0];
    __builtin_prefetch(h2 + (long)rcur * 32, 0, 0);
    __builtin_prefetch(h2 + (long)ccur * 32, 0, 0);
    long e1 = ((long)wave0 + nwaves) * 16 + m;
    if (e1 >= E) e1 = E - 1;
    int rnxt = row[e1], cnxt = col[e1];
    __builtin_prefetch(h2 + (long)rnxt * 32, 0, 0);
    __builtin_prefetch(h2 + (long)cnxt * 32, 0, 0);

    for (int tile = wave0; tile < ntiles; tile += nwaves) {
        long base = (long)tile * 16;

        // gather current rows (cache-warm from prefetch two iterations ago)
        const v4f* pr4 = (const v4f*)(h2 + (long)rcur * 32);
        const v4f* pc4 = (const v4f*)(h2 + (long)ccur * 32);
        v4f r0 = pr4[k0 >> 2], r1 = pr4[(k0 >> 2) + 1];
        v4f r2 = pr4[(k0 + 16) >> 2], r3 = pr4[((k0 + 16) >> 2) + 1];
        v4f q0 = pc4[k0 >> 2], q1 = pc4[(k0 >> 2) + 1];
        v4f q2 = pc4[(k0 + 16) >> 2], q3 = pc4[((k0 + 16) >> 2) + 1];

        // load idx(t+2) and prefetch its rows (issued after the gathers above,
        // so waiting on the gathers does not wait on these)
        long e2 = ((long)tile + 2 * (long)nwaves) * 16 + m;
        if (e2 >= E) e2 = E - 1;
        int rt2 = row[e2], ct2 = col[e2];
        __builtin_prefetch(h2 + (long)rt2 * 32, 0, 0);
        __builtin_prefetch(h2 + (long)ct2 * 32, 0, 0);

        v16h aR, aC;  // +b2 folded in (h2 = agg2 + b2)
#pragma unroll
        for (int i = 0; i < 4; ++i) {
            aR[i]      = (_Float16)(r0[i] + badd[i]);
            aR[4 + i]  = (_Float16)(r1[i] + badd[4 + i]);
            aR[8 + i]  = (_Float16)(r2[i] + badd[8 + i]);
            aR[12 + i] = (_Float16)(r3[i] + badd[12 + i]);
            aC[i]      = (_Float16)(q0[i] + badd[i]);
            aC[4 + i]  = (_Float16)(q1[i] + badd[4 + i]);
            aC[8 + i]  = (_Float16)(q2[i] + badd[8 + i]);
            aC[12 + i] = (_Float16)(q3[i] + badd[12 + i]);
        }

        v8f acc = {};
        acc = WMMA_F16F32(aR, bR, acc);
        acc = WMMA_F16F32(aC, bC, acc);

        float p[8];
#pragma unroll
        for (int v = 0; v < 8; ++v) {
            float t = fmaxf(acc[v] + bb, 0.0f);
            p[v] = t * w2;
        }
#pragma unroll
        for (int mask = 1; mask < 16; mask <<= 1) {
#pragma unroll
            for (int v = 0; v < 8; ++v) p[v] += __shfl_xor(p[v], mask, 32);
        }

        // every lane now holds all 8 per-edge sums of its half;
        // lanes ncol<8 each write one score (16 coalesced stores per tile)
        int vsel = ncol & 7;
        float pv = p[0];
#pragma unroll
        for (int v = 1; v < 8; ++v) pv = (vsel == v) ? p[v] : pv;
        if (ncol < 8) {
            long ei = base + half * 8 + vsel;
            if (ei < E) out[ei] = 1.0f / (1.0f + expf(-(pv + b2s)));
        }

        rcur = rnxt; ccur = cnxt;
        rnxt = rt2;  cnxt = ct2;
    }
}

extern "C" void kernel_launch(void* const* d_in, const int* in_sizes, int n_in,
                              void* d_out, int out_size, void* d_ws, size_t ws_size,
                              hipStream_t stream) {
    const float* x   = (const float*)d_in[0];
    const int*   ei  = (const int*)d_in[1];
    const float* W1  = (const float*)d_in[2];
    const float* b1  = (const float*)d_in[3];
    const float* W2  = (const float*)d_in[4];
    const float* b2  = (const float*)d_in[5];
    const float* Wc1 = (const float*)d_in[6];
    const float* bc1 = (const float*)d_in[7];
    const float* Wc2 = (const float*)d_in[8];
    const float* bc2 = (const float*)d_in[9];

    const int N = in_sizes[0] / 9;
    const int E = in_sizes[1] / 2;
    const int* row = ei;
    const int* col = ei + E;

    // workspace layout (floats, 128B-aligned sections):
    //   dinv[N] | xh1[64N] (h1f16 overlays) | xh2[32N] | agg2[32N] | W2h | Wc1h
    // conv1's agg1 spans the xh2+agg2 region (64N floats) before conv2 reuses it.
    auto aup = [](size_t v) { return (v + 31) & ~(size_t)31; };
    float* ws = (float*)d_ws;
    size_t o = 0;
    float* dinv = ws + o;  o += aup((size_t)N);
    float* xh1  = ws + o;  o += aup((size_t)64 * N);
    size_t agg1_off = o;
    float* agg1 = ws + agg1_off;
    float* xh2  = ws + agg1_off;                  // 32N
    size_t o2 = agg1_off + aup((size_t)32 * N);
    float* agg2 = ws + o2;                        // 32N (becomes h2 in place)
    o = o2 + aup((size_t)32 * N);
    _Float16* W2h  = (_Float16*)(ws + o); o += aup((size_t)(64 * 32) / 2);
    _Float16* Wc1h = (_Float16*)(ws + o); o += aup((size_t)(64 * 16) / 2);
    _Float16* h1f16 = (_Float16*)xh1;
    float* out = (float*)d_out;

    const int B = 256;
    const int ntiles_n = (N + 15) / 16;
    const int ntiles_e = (E + 15) / 16;

    // normalization
    k_deg_init <<<cdiv_u(N, B), B, 0, stream>>>(dinv, N);
    k_deg_count<<<cdiv_u(E, B), B, 0, stream>>>(row, dinv, E);
    k_dinv     <<<cdiv_u(N, B), B, 0, stream>>>(dinv, N);
    k_wcvt     <<<cdiv_u(64 * 32, B), B, 0, stream>>>(W2, Wc1, W2h, Wc1h);

    // conv1
    k_xw1_selfloop<<<cdiv_u((long long)N * 64, B), B, 0, stream>>>(x, W1, dinv, xh1, agg1, N);
    k_scatter64   <<<cdiv_u((long long)E * 32, B), B, 0, stream>>>(row, col, dinv, xh1, agg1, E);
    k_relu_bias_f16<<<cdiv_u((long long)N * 16, B), B, 0, stream>>>(agg1, b1, h1f16, N * 16);

    // conv2 (WMMA, persistent + pipelined; self-loop fused into epilogue)
    {
        unsigned blocks = umin_u(cdiv_u((long long)ntiles_n * 32, B), 2048u);
        k_gemm2_wmma<<<blocks, B, 0, stream>>>(h1f16, W2h, dinv, xh2, agg2, N, ntiles_n);
    }
    k_scatter32<<<cdiv_u((long long)E * 32, B), B, 0, stream>>>(row, col, dinv, xh2, agg2, E);

    // edge scorer (WMMA, persistent + pipelined; +b2 fused into gather)
    {
        unsigned blocks = umin_u(cdiv_u((long long)ntiles_e * 32, B), 4096u);
        k_edge_wmma<<<blocks, B, 0, stream>>>(agg2, row, col, Wc1h, bc1, Wc2, bc2, b2,
                                              out, E, ntiles_e);
    }
}
